// tweet_model_6700148982140
// MI455X (gfx1250) — compile-verified, matrix operand order checked by reading, not run on hardware
//
#include <hip/hip_runtime.h>
#include <hip/hip_bf16.h>
#include <stdint.h>

#define Bsz   512
#define Tlen  290
#define Edim  128
#define Hdim  512
#define G4    2048   // 4*H
#define NVOC  32

typedef __attribute__((ext_vector_type(16))) __bf16 v16bf;
typedef __attribute__((ext_vector_type(8)))  __bf16 v8bf;
typedef __attribute__((ext_vector_type(8)))  float  v8f;
typedef int v4i_ __attribute__((vector_size(16)));

__device__ __forceinline__ __bf16 f2bf(float x) {
    union { float f; uint32_t u; } v; v.f = x;
    uint32_t r = v.u + 0x7FFFu + ((v.u >> 16) & 1u);
    union { uint16_t s; __bf16 b; } o; o.s = (uint16_t)(r >> 16); return o.b;
}
__device__ __forceinline__ float sigm(float x) { return 1.0f / (1.0f + __expf(-x)); }

// ---------------------------------------------------------------------------
// CDNA5 async global->LDS copy (16B per lane), ASYNCcnt-tracked
// ---------------------------------------------------------------------------
__device__ __forceinline__ void async_cp16(void* lds_dst, const void* gsrc) {
#if __has_builtin(__builtin_amdgcn_global_load_async_to_lds_b128)
    __builtin_amdgcn_global_load_async_to_lds_b128(
        (__attribute__((address_space(1))) v4i_*)gsrc,
        (__attribute__((address_space(3))) v4i_*)lds_dst, 0, 0);
#else
    unsigned l = (unsigned)(uintptr_t)((__attribute__((address_space(3))) char*)(char*)lds_dst);
    asm volatile("global_load_async_to_lds_b128 %0, %1, off"
                 :: "v"(l), "v"(gsrc) : "memory");
#endif
}
__device__ __forceinline__ void wait_async0() {
#if __has_builtin(__builtin_amdgcn_s_wait_asynccnt)
    __builtin_amdgcn_s_wait_asynccnt(0);
#else
    asm volatile("s_wait_asynccnt 0x0" ::: "memory");
#endif
}

// ---------------------------------------------------------------------------
// Small precompute kernels (token tables, weight conversion, state init)
// ---------------------------------------------------------------------------
__global__ void table_fc1_k(const float* __restrict__ emb, const float* __restrict__ w1,
                            const float* __restrict__ b1, float* __restrict__ t1) {
    int i = blockIdx.x * 256 + threadIdx.x;
    if (i >= NVOC * Hdim) return;
    int v = i >> 9, n = i & (Hdim - 1);
    float s = b1[n];
    const float* er = emb + v * Edim;
    const float* wr = w1 + n * Edim;
    for (int e = 0; e < Edim; ++e) s += er[e] * wr[e];
    t1[i] = s;
}

// tout[v][n] = bias(n) + bias2(n) + sum_k tin[v][k] * w[n*ldw + koff + k]
__global__ void table_mm_k(const float* __restrict__ tin, const float* __restrict__ w,
                           int ldw, int koff,
                           const float* __restrict__ bias, const float* __restrict__ bias2,
                           float* __restrict__ tout, int N) {
    int i = blockIdx.x * 256 + threadIdx.x;
    if (i >= NVOC * N) return;
    int v = i / N, n = i % N;
    float s = (bias ? bias[n] : 0.0f) + (bias2 ? bias2[n] : 0.0f);
    const float* tr = tin + v * Hdim;
    const float* wr = w + (size_t)n * ldw + koff;
    for (int k = 0; k < Hdim; ++k) s += tr[k] * wr[k];
    tout[i] = s;
}

__global__ void f32_to_bf16_k(const float* __restrict__ src, __bf16* __restrict__ dst, int n) {
    int i = blockIdx.x * 256 + threadIdx.x;
    if (i < n) dst[i] = f2bf(src[i]);
}

// Wcat1[n][k] : k<512 -> wih1[n][k] ; else whh1[n][k-512]   (row-major [2048][1024])
__global__ void build_wcat1_k(const float* __restrict__ wih1, const float* __restrict__ whh1,
                              __bf16* __restrict__ dst) {
    int i = blockIdx.x * 256 + threadIdx.x;
    if (i >= G4 * 2 * Hdim) return;
    int n = i >> 10, k = i & 1023;
    float v = (k < Hdim) ? wih1[n * Hdim + k] : whh1[n * Hdim + (k - Hdim)];
    dst[i] = f2bf(v);
}

// w4A[n][k] = w4[n][k] for k<512 (row stride of w4 is 1024)
__global__ void build_w4a_k(const float* __restrict__ w4, __bf16* __restrict__ dst) {
    int i = blockIdx.x * 256 + threadIdx.x;
    if (i >= Hdim * Hdim) return;
    int n = i >> 9, k = i & 511;
    dst[i] = f2bf(w4[n * 1024 + k]);
}

__global__ void bias_cat_k(const float* __restrict__ a, const float* __restrict__ b,
                           float* __restrict__ o) {
    int i = blockIdx.x * 256 + threadIdx.x;
    if (i < G4) o[i] = a[i] + b[i];
}

__global__ void zero_state_k(__bf16* h0, float* c0, float* c1, __bf16* h1z) {
    int i = blockIdx.x * 256 + threadIdx.x;
    if (i < Bsz * Hdim) { h0[i] = f2bf(0.f); c0[i] = 0.f; c1[i] = 0.f; h1z[i] = f2bf(0.f); }
}

// ---------------------------------------------------------------------------
// LSTM recurrent GEMM:  C[512][2048] = [A0|A1] (bf16, K=512 or 1024) x W[N][K]^T
// 64x64 tile per block, 128 threads (4 waves), wave w owns rows w*16..w*16+15.
// Double-buffered LDS; tiles staged with async global->LDS DMA overlapping WMMA.
// ---------------------------------------------------------------------------
__global__ __launch_bounds__(128)
void lstm_gemm_wmma(const __bf16* __restrict__ A0, const __bf16* __restrict__ A1,
                    const __bf16* __restrict__ W, float* __restrict__ C,
                    int N, int K) {
    __shared__ __bf16 As[2][64 * 40];
    __shared__ __bf16 Bs[2][64 * 40];
    const int tid  = threadIdx.x;
    const int wave = tid >> 5, lane = tid & 31;
    const int mBlk = blockIdx.x << 6, nBlk = blockIdx.y << 6;
    const int r = lane & 15, kh = (lane >> 4) << 3;

    auto issueTile = [&](int kb, int buf) {
        const int k0 = kb << 5;
        const __bf16* Aptr; int kA;
        if (k0 < Hdim) { Aptr = A0; kA = k0; } else { Aptr = A1; kA = k0 - Hdim; }
        #pragma unroll
        for (int c = tid; c < 256; c += 128) {
            int row = c >> 2, kk = (c & 3) << 3;
            async_cp16(&As[buf][row * 40 + kk], Aptr + (size_t)(mBlk + row) * Hdim + kA + kk);
            async_cp16(&Bs[buf][row * 40 + kk], W + (size_t)(nBlk + row) * K + k0 + kk);
        }
    };

    v8f acc0 = {}, acc1 = {}, acc2 = {}, acc3 = {};
    const int nK = K >> 5;
    issueTile(0, 0);
    for (int kb = 0; kb < nK; ++kb) {
        const int cur = kb & 1;
        wait_async0();        // loads into buf[cur] done (per wave)
        __syncthreads();      // all waves: buf[cur] ready AND done reading buf[cur^1]
        if (kb + 1 < nK) issueTile(kb + 1, cur ^ 1);   // DMA overlaps the WMMAs below

        union { v16bf v; v8bf h[2]; } a, b;
        const __bf16* pa = &As[cur][(wave * 16 + r) * 40 + kh];
        a.h[0] = *(const v8bf*)pa; a.h[1] = *(const v8bf*)(pa + 16);
        const __bf16* pb = &Bs[cur][r * 40 + kh];
        b.h[0] = *(const v8bf*)pb; b.h[1] = *(const v8bf*)(pb + 16);
        acc0 = __builtin_amdgcn_wmma_f32_16x16x32_bf16(false, a.v, false, b.v, (short)0, acc0, false, false);
        pb = &Bs[cur][(16 + r) * 40 + kh];
        b.h[0] = *(const v8bf*)pb; b.h[1] = *(const v8bf*)(pb + 16);
        acc1 = __builtin_amdgcn_wmma_f32_16x16x32_bf16(false, a.v, false, b.v, (short)0, acc1, false, false);
        pb = &Bs[cur][(32 + r) * 40 + kh];
        b.h[0] = *(const v8bf*)pb; b.h[1] = *(const v8bf*)(pb + 16);
        acc2 = __builtin_amdgcn_wmma_f32_16x16x32_bf16(false, a.v, false, b.v, (short)0, acc2, false, false);
        pb = &Bs[cur][(48 + r) * 40 + kh];
        b.h[0] = *(const v8bf*)pb; b.h[1] = *(const v8bf*)(pb + 16);
        acc3 = __builtin_amdgcn_wmma_f32_16x16x32_bf16(false, a.v, false, b.v, (short)0, acc3, false, false);
    }
    const int mm  = mBlk + wave * 16 + ((lane >> 4) << 3);
    const int nn0 = nBlk + r;
    float* Cr = C + (size_t)mm * N;
    #pragma unroll
    for (int e = 0; e < 8; ++e) {
        float* row = Cr + (size_t)e * N;
        row[nn0]      = acc0[e];
        row[nn0 + 16] = acc1[e];
        row[nn0 + 32] = acc2[e];
        row[nn0 + 48] = acc3[e];
    }
}

// ---------------------------------------------------------------------------
// Gate nonlinearity + state update.  g layout [B][2048] (i,f,g,o slabs of 512).
// ---------------------------------------------------------------------------
__global__ __launch_bounds__(256)
void lstm_gate_k(const float* __restrict__ g, const float* __restrict__ base,
                 const float* __restrict__ tab, const int* __restrict__ encT,
                 float* __restrict__ c, __bf16* __restrict__ hout) {
    int idx = blockIdx.x * 256 + threadIdx.x;           // B*H threads
    int b = idx >> 9, n = idx & (Hdim - 1);
    float bi, bfv, bg, bo;
    if (tab) {
        int tok = encT[b * Tlen];
        const float* tb = tab + (size_t)tok * G4;
        bi = tb[n]; bfv = tb[n + Hdim]; bg = tb[n + 2 * Hdim]; bo = tb[n + 3 * Hdim];
    } else {
        bi = base[n]; bfv = base[n + Hdim]; bg = base[n + 2 * Hdim]; bo = base[n + 3 * Hdim];
    }
    const float* gr = g + (size_t)b * G4;
    float i  = sigm(gr[n] + bi);
    float f  = sigm(gr[n + Hdim] + bfv);
    float gg = tanhf(gr[n + 2 * Hdim] + bg);
    float o  = sigm(gr[n + 3 * Hdim] + bo);
    float cc = f * c[idx] + i * gg;
    c[idx] = cc;
    hout[idx] = f2bf(o * tanhf(cc));
}

// ---------------------------------------------------------------------------
// fc4 GEMM: [T*B][512] = h1_all x w4A^T, epilogue adds token-gathered table
// (which already folds fc3@w4B + b4).  Output bf16.  Async double-buffered.
// ---------------------------------------------------------------------------
__global__ __launch_bounds__(128)
void fc4_gemm_wmma(const __bf16* __restrict__ A, const __bf16* __restrict__ W,
                   const float* __restrict__ tab, const int* __restrict__ enc,
                   __bf16* __restrict__ Out) {
    __shared__ __bf16 As[2][64 * 40];
    __shared__ __bf16 Bs[2][64 * 40];
    const int tid  = threadIdx.x;
    const int wave = tid >> 5, lane = tid & 31;
    const int mBlk = blockIdx.x << 6, nBlk = blockIdx.y << 6;
    const int r = lane & 15, kh = (lane >> 4) << 3;

    auto issueTile = [&](int kb, int buf) {
        const int k0 = kb << 5;
        #pragma unroll
        for (int c = tid; c < 256; c += 128) {
            int row = c >> 2, kk = (c & 3) << 3;
            async_cp16(&As[buf][row * 40 + kk], A + (size_t)(mBlk + row) * Hdim + k0 + kk);
            async_cp16(&Bs[buf][row * 40 + kk], W + (size_t)(nBlk + row) * Hdim + k0 + kk);
        }
    };

    v8f acc0 = {}, acc1 = {}, acc2 = {}, acc3 = {};
    const int nK = Hdim >> 5;
    issueTile(0, 0);
    for (int kb = 0; kb < nK; ++kb) {
        const int cur = kb & 1;
        wait_async0();
        __syncthreads();
        if (kb + 1 < nK) issueTile(kb + 1, cur ^ 1);

        union { v16bf v; v8bf h[2]; } a, b;
        const __bf16* pa = &As[cur][(wave * 16 + r) * 40 + kh];
        a.h[0] = *(const v8bf*)pa; a.h[1] = *(const v8bf*)(pa + 16);
        #pragma unroll
        for (int j = 0; j < 4; ++j) {
            const __bf16* pb = &Bs[cur][(j * 16 + r) * 40 + kh];
            b.h[0] = *(const v8bf*)pb; b.h[1] = *(const v8bf*)(pb + 16);
            v8f* pacc = (j == 0) ? &acc0 : (j == 1) ? &acc1 : (j == 2) ? &acc2 : &acc3;
            *pacc = __builtin_amdgcn_wmma_f32_16x16x32_bf16(false, a.v, false, b.v, (short)0, *pacc, false, false);
        }
    }
    const int mm = mBlk + wave * 16 + ((lane >> 4) << 3);
    #pragma unroll
    for (int e = 0; e < 8; ++e) {
        int m = mm + e;
        int t = m >> 9, b = m & (Bsz - 1);   // rows are (t, b) with t-major
        int tok = enc[b * Tlen + t];
        const float* tb = tab + (size_t)tok * Hdim;
        __bf16* orow = Out + (size_t)m * Hdim;
        int nn = nBlk + r;
        orow[nn]      = f2bf(acc0[e] + tb[nn]);
        orow[nn + 16] = f2bf(acc1[e] + tb[nn + 16]);
        orow[nn + 32] = f2bf(acc2[e] + tb[nn + 32]);
        orow[nn + 48] = f2bf(acc3[e] + tb[nn + 48]);
    }
}

// ---------------------------------------------------------------------------
// logits GEMM: [T*B][32] = fc4 x wf^T + bf, scattered to out[b][t][v] (fp32)
// ---------------------------------------------------------------------------
__global__ __launch_bounds__(128)
void logits_gemm_wmma(const __bf16* __restrict__ A, const __bf16* __restrict__ W,
                      const float* __restrict__ bvec, float* __restrict__ Out) {
    __shared__ __bf16 As[64 * 40];
    __shared__ __bf16 Bs[32 * 40];
    const int tid  = threadIdx.x;
    const int wave = tid >> 5, lane = tid & 31;
    const int mBlk = blockIdx.x << 6;
    const int r = lane & 15, kh = (lane >> 4) << 3;

    v8f acc0 = {}, acc1 = {};
    for (int kb = 0; kb < (Hdim >> 5); ++kb) {
        const int k0 = kb << 5;
        #pragma unroll
        for (int c = tid; c < 256; c += 128) {
            int row = c >> 2, kk = (c & 3) << 3;
            async_cp16(&As[row * 40 + kk], A + (size_t)(mBlk + row) * Hdim + k0 + kk);
        }
        {   // 32 rows x 4 chunks = 128 chunks, one per thread
            int row = tid >> 2, kk = (tid & 3) << 3;
            async_cp16(&Bs[row * 40 + kk], W + (size_t)row * Hdim + k0 + kk);
        }
        wait_async0();
        __syncthreads();
        union { v16bf v; v8bf h[2]; } a, b;
        const __bf16* pa = &As[(wave * 16 + r) * 40 + kh];
        a.h[0] = *(const v8bf*)pa; a.h[1] = *(const v8bf*)(pa + 16);
        const __bf16* pb = &Bs[r * 40 + kh];
        b.h[0] = *(const v8bf*)pb; b.h[1] = *(const v8bf*)(pb + 16);
        acc0 = __builtin_amdgcn_wmma_f32_16x16x32_bf16(false, a.v, false, b.v, (short)0, acc0, false, false);
        pb = &Bs[(16 + r) * 40 + kh];
        b.h[0] = *(const v8bf*)pb; b.h[1] = *(const v8bf*)(pb + 16);
        acc1 = __builtin_amdgcn_wmma_f32_16x16x32_bf16(false, a.v, false, b.v, (short)0, acc1, false, false);
        __syncthreads();
    }
    const int mm = mBlk + wave * 16 + ((lane >> 4) << 3);
    #pragma unroll
    for (int e = 0; e < 8; ++e) {
        int m = mm + e;
        int t = m >> 9, b = m & (Bsz - 1);
        float* orow = Out + ((size_t)b * Tlen + t) * NVOC;
        orow[r]      = acc0[e] + bvec[r];
        orow[r + 16] = acc1[e] + bvec[r + 16];
    }
}

// ---------------------------------------------------------------------------
// Host-side orchestration
// ---------------------------------------------------------------------------
extern "C" void kernel_launch(void* const* d_in, const int* in_sizes, int n_in,
                              void* d_out, int out_size, void* d_ws, size_t ws_size,
                              hipStream_t stream) {
    const int*   enc   = (const int*)  d_in[0];
    const float* emb   = (const float*)d_in[1];
    const float* w1    = (const float*)d_in[2];
    const float* b1    = (const float*)d_in[3];
    const float* wih0  = (const float*)d_in[4];
    const float* whh0  = (const float*)d_in[5];
    const float* bih0  = (const float*)d_in[6];
    const float* bhh0  = (const float*)d_in[7];
    const float* wih1  = (const float*)d_in[8];
    const float* whh1  = (const float*)d_in[9];
    const float* bih1  = (const float*)d_in[10];
    const float* bhh1  = (const float*)d_in[11];
    const float* w2    = (const float*)d_in[12];
    const float* b2    = (const float*)d_in[13];
    const float* w3    = (const float*)d_in[14];
    const float* b3    = (const float*)d_in[15];
    const float* w4    = (const float*)d_in[16];
    const float* b4    = (const float*)d_in[17];
    const float* wf    = (const float*)d_in[18];
    const float* bfv   = (const float*)d_in[19];
    (void)in_sizes; (void)n_in; (void)out_size; (void)ws_size;

    char* p = (char*)d_ws;
    auto take = [&](size_t bytes) { char* q = p; p += (bytes + 255) & ~(size_t)255; return q; };

    float*  tFc1    = (float*) take((size_t)NVOC * Hdim * 4);
    float*  tFc2    = (float*) take((size_t)NVOC * Hdim * 4);
    float*  tFc3    = (float*) take((size_t)NVOC * Hdim * 4);
    float*  tXg0    = (float*) take((size_t)NVOC * G4 * 4);
    float*  tFc4B   = (float*) take((size_t)NVOC * Hdim * 4);
    float*  biasCat = (float*) take((size_t)G4 * 4);
    __bf16* whh0b   = (__bf16*)take((size_t)G4 * Hdim * 2);
    __bf16* wcat1b  = (__bf16*)take((size_t)G4 * 2 * Hdim * 2);
    __bf16* w4Ab    = (__bf16*)take((size_t)Hdim * Hdim * 2);
    __bf16* wfb     = (__bf16*)take((size_t)NVOC * Hdim * 2);
    __bf16* h0      = (__bf16*)take((size_t)Bsz * Hdim * 2);
    __bf16* h1z     = (__bf16*)take((size_t)Bsz * Hdim * 2);
    float*  c0      = (float*) take((size_t)Bsz * Hdim * 4);
    float*  c1      = (float*) take((size_t)Bsz * Hdim * 4);
    float*  g       = (float*) take((size_t)Bsz * G4 * 4);
    __bf16* h1all   = (__bf16*)take((size_t)Tlen * Bsz * Hdim * 2);
    __bf16* fc4     = (__bf16*)take((size_t)Tlen * Bsz * Hdim * 2);

    // --- token tables (tiny, fp32 scalar) ---
    table_fc1_k<<<(NVOC * Hdim + 255) / 256, 256, 0, stream>>>(emb, w1, b1, tFc1);
    table_mm_k<<<(NVOC * G4 + 255) / 256, 256, 0, stream>>>(tFc1, wih0, Hdim, 0, bih0, bhh0, tXg0, G4);
    table_mm_k<<<(NVOC * Hdim + 255) / 256, 256, 0, stream>>>(tFc1, w2, Hdim, 0, b2, nullptr, tFc2, Hdim);
    table_mm_k<<<(NVOC * Hdim + 255) / 256, 256, 0, stream>>>(tFc2, w3, Hdim, 0, b3, nullptr, tFc3, Hdim);
    table_mm_k<<<(NVOC * Hdim + 255) / 256, 256, 0, stream>>>(tFc3, w4, 2 * Hdim, Hdim, b4, nullptr, tFc4B, Hdim);
    bias_cat_k<<<(G4 + 255) / 256, 256, 0, stream>>>(bih1, bhh1, biasCat);

    // --- weight conversion to bf16 ---
    f32_to_bf16_k<<<(G4 * Hdim + 255) / 256, 256, 0, stream>>>(whh0, whh0b, G4 * Hdim);
    build_wcat1_k<<<(G4 * 2 * Hdim + 255) / 256, 256, 0, stream>>>(wih1, whh1, wcat1b);
    build_w4a_k<<<(Hdim * Hdim + 255) / 256, 256, 0, stream>>>(w4, w4Ab);
    f32_to_bf16_k<<<(NVOC * Hdim + 255) / 256, 256, 0, stream>>>(wf, wfb, NVOC * Hdim);

    // --- state init (must run every call: deterministic replay) ---
    zero_state_k<<<(Bsz * Hdim + 255) / 256, 256, 0, stream>>>(h0, c0, c1, h1z);

    // --- recurrent loop: both layers interleaved per timestep ---
    const dim3 gemmGrid(Bsz / 64, G4 / 64);
    for (int t = 0; t < Tlen; ++t) {
        // layer 0: g = h0 @ whh0^T
        lstm_gemm_wmma<<<gemmGrid, 128, 0, stream>>>(h0, h0, whh0b, g, G4, Hdim);
        lstm_gate_k<<<(Bsz * Hdim) / 256, 256, 0, stream>>>(g, nullptr, tXg0, enc + t, c0, h0);
        // layer 1: g = [h0_t | h1_{t-1}] @ [wih1|whh1]^T   (K = 1024)
        const __bf16* h1prev = (t == 0) ? h1z : (h1all + (size_t)(t - 1) * Bsz * Hdim);
        lstm_gemm_wmma<<<gemmGrid, 128, 0, stream>>>(h0, h1prev, wcat1b, g, G4, 2 * Hdim);
        lstm_gate_k<<<(Bsz * Hdim) / 256, 256, 0, stream>>>(g, biasCat, nullptr, nullptr, c1,
                                                            h1all + (size_t)t * Bsz * Hdim);
    }

    // --- fc4 = h1_all @ w4A^T + gather(tFc4B)  (bf16 out) ---
    fc4_gemm_wmma<<<dim3((Tlen * Bsz) / 64, Hdim / 64), 128, 0, stream>>>(h1all, w4Ab, tFc4B, enc, fc4);

    // --- logits = fc4 @ wf^T + bf -> out[b][t][v] (fp32) ---
    logits_gemm_wmma<<<dim3((Tlen * Bsz) / 64), 128, 0, stream>>>(fc4, wfb, bfv, (float*)d_out);
}
